// SwinTransformerBlock_37091337568929
// MI455X (gfx1250) — matrix operand order repaired and checked by reading, compiled, not run on hardware
//
#include <hip/hip_runtime.h>
#include <hip/hip_bf16.h>

// ---------------- types for CDNA5 WMMA ----------------
typedef __bf16 bf16_t;
typedef __attribute__((ext_vector_type(16))) __bf16 v16bf;
typedef __attribute__((ext_vector_type(8)))  __bf16 v8bf;
typedef __attribute__((ext_vector_type(8)))  float  v8f;

union ABFrag { v16bf v; v8bf h[2]; };
union Pack8  { unsigned short hs[8]; uint4 q; };

// problem constants
#define Bb   32
#define Hh   64
#define Ww_  64
#define Cc   512
#define NHh  16
#define HD   32
#define AA   64      // window area
#define NW   64      // windows per image
#define NWIN 2048    // B * nW
#define MTOK 131072  // B*H*W tokens
#define FFd  2048
#define METAd 384

__device__ __forceinline__ unsigned short f2bf(float f) {
  union { float f; unsigned u; } x; x.f = f;
  unsigned u = x.u;
  u += 0x7fffu + ((u >> 16) & 1u);
  return (unsigned short)(u >> 16);
}
__device__ __forceinline__ float bf2f(unsigned short h) {
  union { unsigned u; float f; } x; x.u = ((unsigned)h) << 16;
  return x.f;
}
__device__ __forceinline__ v8f vzero8() {
  v8f z;
#pragma unroll
  for (int i = 0; i < 8; ++i) z[i] = 0.f;
  return z;
}

// A-fragment (16-bit A 16x32): lane m = lane&15, half = lane>>4.
// elements 0..7  -> k = koff + 8*half + 0..7      (one b128)
// elements 8..15 -> k = koff + 16 + 8*half + 0..7 (one b128)
__device__ __forceinline__ v16bf load_frag_a(const unsigned short* lds, int row0, int stride, int koff) {
  int lane = threadIdx.x & 31;
  int half = lane >> 4;
  int m = row0 + (lane & 15);
  ABFrag f;
  f.h[0] = *(const v8bf*)(lds + m * stride + koff + 8 * half);
  f.h[1] = *(const v8bf*)(lds + m * stride + koff + 16 + 8 * half);
  return f.v;
}
// B-fragment (16-bit B 32x16): lane n = lane&15, elements 0..7 -> k = koff + 8*half + 0..7, elements 8..15 -> k = koff + 16 + 8*half + 0..7.
__device__ __forceinline__ v16bf load_frag_b(const unsigned short* lds, int row0, int stride, int koff) {
  int lane = threadIdx.x & 31;
  int half = lane >> 4;
  int n = row0 + (lane & 15);
  ABFrag f;
  f.h[0] = *(const v8bf*)(lds + n * stride + koff + 8 * half);
  f.h[1] = *(const v8bf*)(lds + n * stride + koff + 16 + 8 * half);
  return f.v;
}

__device__ __forceinline__ v8f wmma_bf16(v16bf a, v16bf b, v8f c) {
  return __builtin_amdgcn_wmma_f32_16x16x32_bf16(false, a, false, b, (short)0, c, false, false);
}

// ---------------- prep: weight transpose fp32[K][N] -> bf16[N][K] ----------------
__global__ void wt_kernel(const float* __restrict__ W, unsigned short* __restrict__ Wt, int K, int N) {
  size_t idx = (size_t)blockIdx.x * 256 + threadIdx.x;
  if (idx >= (size_t)K * N) return;
  int n = (int)(idx / K), kk = (int)(idx % K);
  Wt[idx] = f2bf(W[(size_t)kk * N + n]);
}

// ---------------- prep: meta MLP -> bias[NH][A][A] ----------------
__global__ __launch_bounds__(128) void meta_bias_kernel(const float* __restrict__ w1, const float* __restrict__ b1,
                                                        const float* __restrict__ w2, const float* __restrict__ b2,
                                                        float* __restrict__ bias) {
  int pair = blockIdx.x;           // i*64 + j
  int i = pair >> 6, j = pair & 63;
  float dy = (float)((i >> 3) - (j >> 3));
  float dx = (float)((i & 7) - (j & 7));
  float r0 = (dy > 0.f ? 1.f : (dy < 0.f ? -1.f : 0.f)) * log1pf(fabsf(dy));
  float r1 = (dx > 0.f ? 1.f : (dx < 0.f ? -1.f : 0.f)) * log1pf(fabsf(dx));
  __shared__ float hid[METAd];
  for (int u = threadIdx.x; u < METAd; u += 128) {
    float h = r0 * w1[u] + r1 * w1[METAd + u] + b1[u];
    hid[u] = h > 0.f ? h : 0.f;
  }
  __syncthreads();
  if (threadIdx.x < NHh) {
    float acc = b2[threadIdx.x];
    for (int u = 0; u < METAd; ++u) acc += hid[u] * w2[u * NHh + threadIdx.x];
    bias[((size_t)threadIdx.x << 12) + pair] = acc;
  }
}

// ---------------- prep: shifted-window mask[nW][A][A] ----------------
__device__ __forceinline__ int regid(int p) { return p < 56 ? 0 : (p < 60 ? 1 : 2); }
__global__ void mask_kernel(float* __restrict__ mask) {
  int idx = blockIdx.x * 256 + threadIdx.x;   // 64*64*64
  int w = idx >> 12, i = (idx >> 6) & 63, j = idx & 63;
  int wy = w >> 3, wx = w & 7;
  int ci = regid(wy * 8 + (i >> 3)) * 3 + regid(wx * 8 + (i & 7));
  int cj = regid(wy * 8 + (j >> 3)) * 3 + regid(wx * 8 + (j & 7));
  mask[idx] = (ci != cj) ? -100.f : 0.f;
}

// ---------------- prep: roll(-4,-4) + window partition, fp32 -> bf16 ----------------
__global__ __launch_bounds__(256) void xw_kernel(const float* __restrict__ x, unsigned short* __restrict__ xw) {
  size_t idx = (size_t)blockIdx.x * 256 + threadIdx.x;  // MTOK * (C/8)
  int c0 = (int)(idx & 63) * 8;
  size_t m = idx >> 6;
  int a = (int)(m & 63);
  size_t win = m >> 6;
  int wi = (int)(win & 63), b = (int)(win >> 6);
  int wy = wi >> 3, wx = wi & 7, iy = a >> 3, ix = a & 7;
  int h = (wy * 8 + iy + 4) & 63, w = (wx * 8 + ix + 4) & 63;
  const float* src = x + (((size_t)b * 4096) + h * 64 + w) * Cc + c0;
  float4 t0 = *(const float4*)src;
  float4 t1 = *(const float4*)(src + 4);
  Pack8 u;
  u.hs[0] = f2bf(t0.x); u.hs[1] = f2bf(t0.y); u.hs[2] = f2bf(t0.z); u.hs[3] = f2bf(t0.w);
  u.hs[4] = f2bf(t1.x); u.hs[5] = f2bf(t1.y); u.hs[6] = f2bf(t1.z); u.hs[7] = f2bf(t1.w);
  *(uint4*)&xw[m * Cc + c0] = u.q;
}

// ---------------- main tiled WMMA GEMM: C[M,N] = A[M,K] @ Wt[N,K]^T + bias ----------------
// block = 256 threads (8 waves, 2x4), block tile 128x128, K-step 32.
// Each wave owns 64x32 (4x2 wmma tiles, 8 WMMAs / K-step, 64 accum VGPRs).
// Double-buffered LDS + software pipeline: global loads for chunk k+1 are in
// flight while chunk k runs through the WMMA pipe; one barrier per K-step.
// EPI: 0 = QKV scatter, 1 = proj + window-reverse, 2 = MLP1 + GELU -> bf16, 3 = MLP2 -> f32
template <int EPI>
__global__ __launch_bounds__(256) void gemm_bf16_kernel(const unsigned short* __restrict__ A,
                                                        const unsigned short* __restrict__ Wt,
                                                        const float* __restrict__ bias,
                                                        void* __restrict__ out0, void* __restrict__ out1,
                                                        void* __restrict__ out2, int M, int N, int K) {
  __shared__ __align__(16) unsigned short As[2][128 * 40];
  __shared__ __align__(16) unsigned short Bs[2][128 * 40];
  const int tid = threadIdx.x, lane = tid & 31, wave = tid >> 5;
  const int wm = wave & 1, wn = wave >> 1;       // 2 x 4 wave grid
  const int m0 = blockIdx.y * 128, n0 = blockIdx.x * 128;
  const int half = lane >> 4;

  v8f acc[4][2];
#pragma unroll
  for (int mt = 0; mt < 4; ++mt) { acc[mt][0] = vzero8(); acc[mt][1] = vzero8(); }

  // tile loader: 128 rows x 32 k per tile, 16 bf16 per thread (two b128)
  const int lr = tid >> 1, lkq = (tid & 1) * 16;
  const unsigned short* Aptr = A + (size_t)(m0 + lr) * K + lkq;
  const unsigned short* Bptr = Wt + (size_t)(n0 + lr) * K + lkq;
  const int lds_off = lr * 40 + lkq;

  uint4 a0 = *(const uint4*)(Aptr);
  uint4 a1 = *(const uint4*)(Aptr + 8);
  uint4 b0 = *(const uint4*)(Bptr);
  uint4 b1 = *(const uint4*)(Bptr + 8);
  *(uint4*)&As[0][lds_off] = a0;  *(uint4*)&As[0][lds_off + 8] = a1;
  *(uint4*)&Bs[0][lds_off] = b0;  *(uint4*)&Bs[0][lds_off + 8] = b1;
  __syncthreads();

  int cur = 0;
  for (int kc = 0; kc < K; kc += 32) {
    const bool more = (kc + 32) < K;
    if (more) {  // stage next K-chunk while WMMAs below execute
      a0 = *(const uint4*)(Aptr + kc + 32);
      a1 = *(const uint4*)(Aptr + kc + 40);
      b0 = *(const uint4*)(Bptr + kc + 32);
      b1 = *(const uint4*)(Bptr + kc + 40);
    }
    v16bf af[4], bfr[2];
#pragma unroll
    for (int mt = 0; mt < 4; ++mt) af[mt] = load_frag_a(&As[cur][0], wm * 64 + mt * 16, 40, 0);
#pragma unroll
    for (int nt = 0; nt < 2; ++nt) bfr[nt] = load_frag_b(&Bs[cur][0], wn * 32 + nt * 16, 40, 0);
#pragma unroll
    for (int mt = 0; mt < 4; ++mt)
#pragma unroll
      for (int nt = 0; nt < 2; ++nt)
        acc[mt][nt] = wmma_bf16(af[mt], bfr[nt], acc[mt][nt]);
    if (more) {
      int nxt = cur ^ 1;
      *(uint4*)&As[nxt][lds_off] = a0;  *(uint4*)&As[nxt][lds_off + 8] = a1;
      *(uint4*)&Bs[nxt][lds_off] = b0;  *(uint4*)&Bs[nxt][lds_off + 8] = b1;
      __syncthreads();
      cur = nxt;
    }
  }

#pragma unroll
  for (int mt = 0; mt < 4; ++mt)
#pragma unroll
    for (int nt = 0; nt < 2; ++nt)
#pragma unroll
      for (int r = 0; r < 8; ++r) {
        int gm = m0 + wm * 64 + mt * 16 + r + 8 * half;
        int gn = n0 + wn * 32 + nt * 16 + (lane & 15);
        float val = acc[mt][nt][r] + bias[gn];
        if (EPI == 0) {  // QKV: n = qkvi*512 + head*32 + d ; m = win*64 + a
          int qkvi = gn >> 9, rem = gn & 511;
          int head = rem >> 5, d = rem & 31;
          int win = gm >> 6, a = gm & 63;
          size_t wh = (size_t)win * NHh + head;
          unsigned short hv = f2bf(val);
          if (qkvi == 0)      ((unsigned short*)out0)[(wh * AA + a) * HD + d] = hv;  // q [wh][a][d]
          else if (qkvi == 1) ((unsigned short*)out1)[(wh * AA + a) * HD + d] = hv;  // k [wh][a][d]
          else                ((unsigned short*)out2)[(wh * HD + d) * AA + a] = hv;  // v^T [wh][d][a]
        } else if (EPI == 1) {  // proj -> window-reverse + roll(+4,+4), f32 token-major
          int win = gm >> 6, a = gm & 63;
          int b = win >> 6, wi = win & 63;
          int wy = wi >> 3, wx = wi & 7, iy = a >> 3, ix = a & 7;
          int h = (wy * 8 + iy + 4) & 63, w = (wx * 8 + ix + 4) & 63;
          ((float*)out0)[(((size_t)b * 4096) + h * 64 + w) * Cc + gn] = val;
        } else if (EPI == 2) {  // MLP1 + exact GELU -> bf16
          float g = 0.5f * val * (1.f + erff(val * 0.70710678118654752f));
          ((unsigned short*)out0)[(size_t)gm * FFd + gn] = f2bf(g);
        } else {                // MLP2 -> f32 token-major
          ((float*)out0)[(size_t)gm * Cc + gn] = val;
        }
      }
}

// ---------------- attention: one block per (window, head), 128 threads ----------------
__global__ __launch_bounds__(128) void attn_kernel(const unsigned short* __restrict__ q,
                                                   const unsigned short* __restrict__ k,
                                                   const unsigned short* __restrict__ vT,
                                                   unsigned short* __restrict__ ctx,
                                                   const float* __restrict__ bias,
                                                   const float* __restrict__ mask,
                                                   const float* __restrict__ tau) {
  const int wh = blockIdx.x;
  const int win = wh >> 4, head = wh & 15, wimg = win & 63;
  const int tid = threadIdx.x, lane = tid & 31, wv = tid >> 5, half = lane >> 4;
  __shared__ __align__(16) unsigned short Qs[64 * 40];
  __shared__ __align__(16) unsigned short Ks[64 * 40];
  __shared__ __align__(16) unsigned short Vs[32 * 72];
  __shared__ __align__(16) unsigned short Ps[64 * 72];
  __shared__ float qn[64], kn[64];

  const size_t base = (size_t)wh * (AA * HD);
  {
    int r = tid >> 1, kq = (tid & 1) * 16;
    *(uint4*)&Qs[r * 40 + kq]     = *(const uint4*)&q[base + r * HD + kq];
    *(uint4*)&Qs[r * 40 + kq + 8] = *(const uint4*)&q[base + r * HD + kq + 8];
    *(uint4*)&Ks[r * 40 + kq]     = *(const uint4*)&k[base + r * HD + kq];
    *(uint4*)&Ks[r * 40 + kq + 8] = *(const uint4*)&k[base + r * HD + kq + 8];
    int d = tid >> 2, aq = (tid & 3) * 16;
    *(uint4*)&Vs[d * 72 + aq]     = *(const uint4*)&vT[base + d * AA + aq];
    *(uint4*)&Vs[d * 72 + aq + 8] = *(const uint4*)&vT[base + d * AA + aq + 8];
  }
  __syncthreads();
  {  // cosine-attention row norms
    int rr = tid & 63;
    const unsigned short* src = (tid < 64) ? &Qs[rr * 40] : &Ks[rr * 40];
    float s = 0.f;
#pragma unroll
    for (int u = 0; u < HD; ++u) { float f = bf2f(src[u]); s += f * f; }
    if (tid < 64) qn[rr] = sqrtf(s); else kn[rr] = sqrtf(s);
  }
  __syncthreads();
  const float tauc = fmaxf(tau[head], 0.01f);

  // S = q @ k^T  (each wave: rows wv*16..+15, all 64 cols)
  v16bf aq_ = load_frag_a(Qs, wv * 16, 40, 0);
  v8f sacc[4];
#pragma unroll
  for (int nt = 0; nt < 4; ++nt) {
    sacc[nt] = vzero8();
    sacc[nt] = wmma_bf16(aq_, load_frag_b(Ks, nt * 16, 40, 0), sacc[nt]);
  }

  float sv[4][8];
#pragma unroll
  for (int nt = 0; nt < 4; ++nt)
#pragma unroll
    for (int r = 0; r < 8; ++r) {
      int i = wv * 16 + r + 8 * half;
      int j = nt * 16 + (lane & 15);
      float den = fmaxf(qn[i] * kn[j], 1e-6f);
      sv[nt][r] = sacc[nt][r] / (den * tauc)
                + bias[((size_t)head << 12) + i * 64 + j]
                + mask[((size_t)wimg << 12) + i * 64 + j];
    }

  // softmax per row: row i lives across 4 n-tiles x 16 lanes of the same half
#pragma unroll
  for (int r = 0; r < 8; ++r) {
    float mx = fmaxf(fmaxf(sv[0][r], sv[1][r]), fmaxf(sv[2][r], sv[3][r]));
    for (int msk = 1; msk < 16; msk <<= 1) mx = fmaxf(mx, __shfl_xor(mx, msk));
    float sm = 0.f;
#pragma unroll
    for (int nt = 0; nt < 4; ++nt) { sv[nt][r] = __expf(sv[nt][r] - mx); sm += sv[nt][r]; }
    for (int msk = 1; msk < 16; msk <<= 1) sm += __shfl_xor(sm, msk);
    float inv = 1.f / sm;
    int i = wv * 16 + r + 8 * half;
#pragma unroll
    for (int nt = 0; nt < 4; ++nt) {
      int j = nt * 16 + (lane & 15);
      Ps[i * 72 + j] = f2bf(sv[nt][r] * inv);
    }
  }
  __syncthreads();

  // O = P @ V : A = P rows wv*16..+15 (K=64 in two steps), B = v^T (n = head-dim)
  v8f oacc[2];
  oacc[0] = vzero8(); oacc[1] = vzero8();
#pragma unroll
  for (int kt = 0; kt < 2; ++kt) {
    v16bf ap = load_frag_a(Ps, wv * 16, 72, kt * 32);
    oacc[0] = wmma_bf16(ap, load_frag_b(Vs, 0, 72, kt * 32), oacc[0]);
    oacc[1] = wmma_bf16(ap, load_frag_b(Vs, 16, 72, kt * 32), oacc[1]);
  }
#pragma unroll
  for (int nt = 0; nt < 2; ++nt)
#pragma unroll
    for (int r = 0; r < 8; ++r) {
      int i = wv * 16 + r + 8 * half;
      int d = nt * 16 + (lane & 15);
      ctx[(((size_t)win << 6) + i) * Cc + (head << 5) + d] = f2bf(oacc[nt][r]);
    }
}

// ---------------- wave-per-token LayerNorm + residual ----------------
__global__ __launch_bounds__(256) void ln_res_kernel(const float* __restrict__ resid,
                                                     const float* __restrict__ y,
                                                     const float* __restrict__ g,
                                                     const float* __restrict__ bta,
                                                     float* __restrict__ of,
                                                     unsigned short* __restrict__ ob) {
  int lane = threadIdx.x & 31;
  size_t token = (size_t)blockIdx.x * 8 + (threadIdx.x >> 5);
  const float* row = y + token * Cc;
  int c0 = lane * 16;
  float v[16];
  float s = 0.f, ss = 0.f;
#pragma unroll
  for (int u = 0; u < 4; ++u) {
    float4 t = *(const float4*)(row + c0 + u * 4);
    v[4 * u + 0] = t.x; v[4 * u + 1] = t.y; v[4 * u + 2] = t.z; v[4 * u + 3] = t.w;
  }
#pragma unroll
  for (int u = 0; u < 16; ++u) { s += v[u]; ss += v[u] * v[u]; }
  for (int msk = 1; msk < 32; msk <<= 1) { s += __shfl_xor(s, msk); ss += __shfl_xor(ss, msk); }
  float mu = s * (1.f / Cc);
  float var = ss * (1.f / Cc) - mu * mu;
  float rs = rsqrtf(var + 1e-5f);
  const float* rrow = resid + token * Cc;
  float o[16];
#pragma unroll
  for (int u = 0; u < 4; ++u) {
    float4 t = *(const float4*)(rrow + c0 + u * 4);
    o[4 * u + 0] = t.x; o[4 * u + 1] = t.y; o[4 * u + 2] = t.z; o[4 * u + 3] = t.w;
  }
#pragma unroll
  for (int u = 0; u < 16; ++u) {
    int c = c0 + u;
    o[u] = o[u] + (v[u] - mu) * rs * g[c] + bta[c];
  }
#pragma unroll
  for (int u = 0; u < 4; ++u) {
    float4 t; t.x = o[4 * u]; t.y = o[4 * u + 1]; t.z = o[4 * u + 2]; t.w = o[4 * u + 3];
    *(float4*)(of + token * Cc + c0 + u * 4) = t;
  }
  if (ob) {
    Pack8 p0, p1;
#pragma unroll
    for (int u = 0; u < 8; ++u) { p0.hs[u] = f2bf(o[u]); p1.hs[u] = f2bf(o[8 + u]); }
    *(uint4*)&ob[token * Cc + c0] = p0.q;
    *(uint4*)&ob[token * Cc + c0 + 8] = p1.q;
  }
}

// ---------------- host side ----------------
extern "C" void kernel_launch(void* const* d_in, const int* in_sizes, int n_in,
                              void* d_out, int out_size, void* d_ws, size_t ws_size,
                              hipStream_t stream) {
  const float* x      = (const float*)d_in[0];
  const float* qkv_w  = (const float*)d_in[1];
  const float* qkv_b  = (const float*)d_in[2];
  const float* proj_w = (const float*)d_in[3];
  const float* proj_b = (const float*)d_in[4];
  const float* tau    = (const float*)d_in[5];
  const float* mw1    = (const float*)d_in[6];
  const float* mb1    = (const float*)d_in[7];
  const float* mw2    = (const float*)d_in[8];
  const float* mb2    = (const float*)d_in[9];
  const float* n1g    = (const float*)d_in[10];
  const float* n1b    = (const float*)d_in[11];
  const float* n2g    = (const float*)d_in[12];
  const float* n2b    = (const float*)d_in[13];
  const float* mlp_w1 = (const float*)d_in[14];
  const float* mlp_b1 = (const float*)d_in[15];
  const float* mlp_w2 = (const float*)d_in[16];
  const float* mlp_b2 = (const float*)d_in[17];

  char* p = (char*)d_ws;
  size_t off = 0;
  auto alloc = [&](size_t bytes) { char* r = p + off; off += (bytes + 255) & ~(size_t)255; return r; };
  unsigned short* WqkvT  = (unsigned short*)alloc((size_t)1536 * 512 * 2);
  unsigned short* WprojT = (unsigned short*)alloc((size_t)512 * 512 * 2);
  unsigned short* Wm1T   = (unsigned short*)alloc((size_t)2048 * 512 * 2);
  unsigned short* Wm2T   = (unsigned short*)alloc((size_t)512 * 2048 * 2);
  float*          biasT  = (float*)alloc((size_t)16 * 4096 * 4);
  float*          maskB  = (float*)alloc((size_t)64 * 4096 * 4);
  unsigned short* xw     = (unsigned short*)alloc((size_t)MTOK * Cc * 2);   // also reused as ctx
  unsigned short* qb     = (unsigned short*)alloc((size_t)MTOK * Cc * 2);
  unsigned short* kb     = (unsigned short*)alloc((size_t)MTOK * Cc * 2);
  unsigned short* vT     = (unsigned short*)alloc((size_t)MTOK * Cc * 2);
  float*          outbuf = (float*)alloc((size_t)MTOK * Cc * 4);            // proj out, then mlp2 out
  float*          x1f    = (float*)alloc((size_t)MTOK * Cc * 4);
  unsigned short* x1b    = (unsigned short*)alloc((size_t)MTOK * Cc * 2);
  unsigned short* hmid   = (unsigned short*)alloc((size_t)MTOK * FFd * 2);
  unsigned short* ctx    = xw;  // xw dead after QKV GEMM

  // 1) weight transposes (fp32 [K][N] -> bf16 [N][K])
  wt_kernel<<<(1536 * 512 + 255) / 256, 256, 0, stream>>>(qkv_w, WqkvT, 512, 1536);
  wt_kernel<<<(512 * 512 + 255) / 256, 256, 0, stream>>>(proj_w, WprojT, 512, 512);
  wt_kernel<<<(512 * 2048 + 255) / 256, 256, 0, stream>>>(mlp_w1, Wm1T, 512, 2048);
  wt_kernel<<<(2048 * 512 + 255) / 256, 256, 0, stream>>>(mlp_w2, Wm2T, 2048, 512);
  // 2) relative-position bias + shift mask
  meta_bias_kernel<<<4096, 128, 0, stream>>>(mw1, mb1, mw2, mb2, biasT);
  mask_kernel<<<(64 * 4096) / 256, 256, 0, stream>>>(maskB);
  // 3) shifted-window partition of x, fp32 -> bf16
  xw_kernel<<<(MTOK * (Cc / 8)) / 256, 256, 0, stream>>>(x, xw);
  // 4) QKV GEMM (M=131072, N=1536, K=512) with head-split scatter epilogue
  gemm_bf16_kernel<0><<<dim3(1536 / 128, MTOK / 128), 256, 0, stream>>>(
      xw, WqkvT, qkv_b, qb, kb, vT, MTOK, 1536, 512);
  // 5) cosine attention per (window, head)
  attn_kernel<<<NWIN * NHh, 128, 0, stream>>>(qb, kb, vT, ctx, biasT, maskB, tau);
  // 6) proj GEMM + window-reverse/unshift (M=131072, N=512, K=512)
  gemm_bf16_kernel<1><<<dim3(512 / 128, MTOK / 128), 256, 0, stream>>>(
      ctx, WprojT, proj_b, outbuf, nullptr, nullptr, MTOK, 512, 512);
  // 7) x1 = x + LN(attn_out)
  ln_res_kernel<<<MTOK / 8, 256, 0, stream>>>(x, outbuf, n1g, n1b, x1f, x1b);
  // 8) MLP1 + GELU (M=131072, N=2048, K=512)
  gemm_bf16_kernel<2><<<dim3(2048 / 128, MTOK / 128), 256, 0, stream>>>(
      x1b, Wm1T, mlp_b1, hmid, nullptr, nullptr, MTOK, 2048, 512);
  // 9) MLP2 (M=131072, N=512, K=2048)
  gemm_bf16_kernel<3><<<dim3(512 / 128, MTOK / 128), 256, 0, stream>>>(
      hmid, Wm2T, mlp_b2, outbuf, nullptr, nullptr, MTOK, 512, 2048);
  // 10) out = x1 + LN(hmid2)
  ln_res_kernel<<<MTOK / 8, 256, 0, stream>>>(x1f, outbuf, n2g, n2b, (float*)d_out, nullptr);
  (void)in_sizes; (void)n_in; (void)out_size; (void)ws_size;
}
